// DotAttention_75368086110276
// MI455X (gfx1250) — matrix operand order repaired and checked by reading, not compiled, not run on hardware
//
#include <hip/hip_runtime.h>
#include <math.h>

typedef __bf16 bf16_t;
typedef __attribute__((ext_vector_type(16))) __bf16 v16bf;
typedef __attribute__((ext_vector_type(8)))  __bf16 v8bf;
typedef __attribute__((ext_vector_type(4)))  __bf16 v4bf;
typedef __attribute__((ext_vector_type(8)))  float  v8f;
typedef __attribute__((ext_vector_type(4)))  float  v4f;

#define DIM     1024
#define T_ENC   1024
#define T_DEC   1024
#define NBATCH  16
#define MROWS   64             // decoder rows per workgroup (K/V reuse factor)
#define QP      (DIM + 8)      // bf16 row stride: +16B pad rotates LDS banks
#define SCALE   (1.0f / 32.0f) // 1/sqrt(1024)

#define SMEM_QP_BYTES  (MROWS * QP * 2)                 // 132096: Q, then P (bf16)
#define SMEM_TOTAL     (SMEM_QP_BYTES + MROWS * 4 + MROWS * 4)

// monotone float <-> uint key (for ds_max_u32-based row max)
__device__ __forceinline__ unsigned fkey(float x) {
    unsigned u = __float_as_uint(x);
    return (u & 0x80000000u) ? ~u : (u | 0x80000000u);
}
__device__ __forceinline__ float funkey(unsigned k) {
    unsigned u = (k & 0x80000000u) ? (k & 0x7fffffffu) : ~k;
    return __uint_as_float(u);
}

// ---- A fragment (16x32 bf16, row-major bf16 source in LDS) ----
// layout: lane r=lane&15, hi=lane>>4; half h -> K = h + 16*hi (contiguous, matches B)
__device__ __forceinline__ v16bf load_A_lds(const bf16_t* row_k0, int hi) {
    const v8bf lo = *(const v8bf*)(row_k0 + 16 * hi);
    const v8bf hh = *(const v8bf*)(row_k0 + 16 * hi + 8);
    return __builtin_shufflevector(lo, hh, 0,1,2,3,4,5,6,7,8,9,10,11,12,13,14,15);
}

// ---- B fragment (32x16) from pre-converted bf16, contiguous along k ------
// lane n=lane&15, hi=lane>>4; half h -> K = h + 16*hi; src is 32B-aligned
__device__ __forceinline__ v16bf load_B_bf16(const bf16_t* src) {
    return *(const v16bf*)src;                 // lowers to 2x global_load_b128
}

// ---- B fragment (32x16) built from fp32 K rows (fallback path) -----------
__device__ __forceinline__ v16bf load_B_f32rows(const float* src) {
    v16bf b;
    const v4f* p = (const v4f*)src;
#pragma unroll
    for (int q = 0; q < 4; ++q) {
        v4f f = p[q];
#pragma unroll
        for (int j = 0; j < 4; ++j) b[q * 4 + j] = (bf16_t)f[j];
    }
    return b;
}

// ---- B fragment from fp32 V with k-stride D (fallback path) --------------
__device__ __forceinline__ v16bf load_B_f32strided(const float* src) {
    v16bf b;
#pragma unroll
    for (int h = 0; h < 16; ++h) b[h] = (bf16_t)src[(size_t)h * DIM];
    return b;
}

// ============================ pre-pass kernels ============================
__global__ __launch_bounds__(256, 4)
void convert_k_bf16(const float* __restrict__ K, bf16_t* __restrict__ Kbf) {
    const size_t i = ((size_t)blockIdx.x * 256 + threadIdx.x) * 4;
    v4f f = __builtin_nontemporal_load((const v4f*)(K + i));
    v4bf o;
#pragma unroll
    for (int j = 0; j < 4; ++j) o[j] = (bf16_t)f[j];
    *(v4bf*)(Kbf + i) = o;                     // RT store: read back soon (L2)
}

#define TP 68   // bf16 tile stride: keeps 8B alignment, spreads banks
__global__ __launch_bounds__(256, 2)
void transpose_v_bf16(const float* __restrict__ V, bf16_t* __restrict__ VT) {
    __shared__ bf16_t tile[64 * TP];
    const int b  = blockIdx.z;
    const int t0 = blockIdx.y * 64;            // encoder-time tile
    const int d0 = blockIdx.x * 64;            // feature tile
    const float* Vb  = V  + ((size_t)b * T_ENC + t0) * DIM + d0;
    bf16_t*      VTb = VT + ((size_t)b * DIM + d0) * T_ENC + t0;
    const int t = threadIdx.x;
#pragma unroll
    for (int i = 0; i < 4; ++i) {
        const int idx = t + i * 256;
        const int r = idx >> 4, c4 = (idx & 15) * 4;
        v4f f = __builtin_nontemporal_load((const v4f*)(Vb + (size_t)r * DIM + c4));
        v4bf q;
#pragma unroll
        for (int j = 0; j < 4; ++j) q[j] = (bf16_t)f[j];
        *(v4bf*)(tile + r * TP + c4) = q;
    }
    __syncthreads();
#pragma unroll
    for (int i = 0; i < 4; ++i) {
        const int idx = t + i * 256;
        const int r = idx >> 4, c4 = (idx & 15) * 4;
        v4bf q;
#pragma unroll
        for (int j = 0; j < 4; ++j) q[j] = tile[(c4 + j) * TP + r];
        *(v4bf*)(VTb + (size_t)r * T_ENC + c4) = q;   // RT store: L2-resident
    }
}

// ============================== main kernel ===============================
template <bool PRECONV>
__global__ __launch_bounds__(256, 1)
void DotAttention_75368086110276_kernel(const float* __restrict__ Q,
                                        const float* __restrict__ K,
                                        const float* __restrict__ V,
                                        const bf16_t* __restrict__ Kbf,
                                        const bf16_t* __restrict__ VTbf,
                                        float* __restrict__ summaries,
                                        float* __restrict__ scores) {
    extern __shared__ __align__(16) char smem[];
    bf16_t*   ldsQ      = (bf16_t*)smem;                            // 64 x QP (later P)
    unsigned* rowMaxKey = (unsigned*)(smem + SMEM_QP_BYTES);        // [64]
    float*    rowInv    = (float*)(smem + SMEM_QP_BYTES + MROWS*4); // [64]

    const int tid  = threadIdx.x;
    const int wave = tid >> 5;
    const int lane = tid & 31;
    const int ln16 = lane & 15;
    const int hi   = lane >> 4;

    const int batch = blockIdx.x >> 4;            // 16 strips per batch
    const int tb    = (blockIdx.x & 15) * MROWS;  // first decoder row of strip

    const float*  Qb  = Q + ((size_t)batch * T_DEC) * DIM;
    const float*  Kb  = K + ((size_t)batch * T_ENC) * DIM;
    const float*  Vb  = V + ((size_t)batch * T_ENC) * DIM;
    const bf16_t* Kbb = PRECONV ? Kbf  + ((size_t)batch * T_ENC) * DIM : nullptr;
    const bf16_t* VTb = PRECONV ? VTbf + ((size_t)batch * DIM) * T_ENC : nullptr;
    float* scB = scores + ((size_t)batch * T_DEC + tb) * T_ENC;

    if (tid < MROWS) rowMaxKey[tid] = 0u;   // key(-inf)

    // -------- Phase 0: stage Q strip -> LDS bf16 (coalesced, streaming) ----
#pragma unroll 4
    for (int chunk = tid; chunk < MROWS * (DIM / 4); chunk += 256) {
        const int r  = chunk >> 8;               // 256 float4 per row
        const int c4 = (chunk & 255) * 4;
        v4f f = __builtin_nontemporal_load(
                    (const v4f*)(Qb + (size_t)(tb + r) * DIM + c4));
        v4bf q;
#pragma unroll
        for (int j = 0; j < 4; ++j) q[j] = (bf16_t)f[j];
        *(v4bf*)(ldsQ + r * QP + c4) = q;
    }
    __syncthreads();

    // -------- Phase 1: raw S = scale*Q K^T -> scores buffer; row max -------
    const int mtile = wave & 3;        // 4 m-tiles of 16 rows
    const int ngrp  = wave >> 2;       // 2 n-groups of 32 n-tiles
    const int mbase = mtile * 16;
    {
        const bf16_t* arow = ldsQ + (mbase + ln16) * QP;
        v8f rmax;
#pragma unroll
        for (int v = 0; v < 8; ++v) rmax[v] = -__builtin_inff();

        for (int nt = ngrp * 32; nt < ngrp * 32 + 32; ++nt) {
            const int nb = nt * 16;
            v8f acc = {};
            if constexpr (PRECONV) {
                const bf16_t* brow = Kbb + (size_t)(nb + ln16) * DIM + 16 * hi;
#pragma unroll 2
                for (int k0 = 0; k0 < DIM; k0 += 32)
                    acc = __builtin_amdgcn_wmma_f32_16x16x32_bf16(
                            false, load_A_lds(arow + k0, hi),
                            false, load_B_bf16(brow + k0),
                            (short)0, acc, false, false);
            } else {
                const float* brow = Kb + (size_t)(nb + ln16) * DIM + 16 * hi;
#pragma unroll 2
                for (int k0 = 0; k0 < DIM; k0 += 32)
                    acc = __builtin_amdgcn_wmma_f32_16x16x32_bf16(
                            false, load_A_lds(arow + k0, hi),
                            false, load_B_f32rows(brow + k0),
                            (short)0, acc, false, false);
            }
#pragma unroll
            for (int v = 0; v < 8; ++v) {
                const float s = acc[v] * SCALE;
                rmax[v] = fmaxf(rmax[v], s);
                scB[(size_t)(mbase + v + 8 * hi) * T_ENC + nb + ln16] = s;  // raw (L2)
            }
        }
#pragma unroll
        for (int v = 0; v < 8; ++v) {
            float x = rmax[v];
            x = fmaxf(x, __shfl_xor(x, 1, 32));
            x = fmaxf(x, __shfl_xor(x, 2, 32));
            x = fmaxf(x, __shfl_xor(x, 4, 32));
            x = fmaxf(x, __shfl_xor(x, 8, 32));
            if (ln16 == 0) atomicMax(&rowMaxKey[mbase + v + 8 * hi], fkey(x));
        }
    }
    __threadfence();     // raw S visible through L2 before read-back
    __syncthreads();

    // -------- Phase 2a: rowwise exp-sum (read raw S back from L2) ----------
    {
        const int r = tid >> 2;                  // 4 threads per row
        const int j = tid & 3;
        const float m = funkey(rowMaxKey[r]);
        const float* srow = scB + (size_t)r * T_ENC;
        float s = 0.0f;
#pragma unroll 4
        for (int i = 0; i < 64; ++i) {
            v4f f = *(const v4f*)(srow + (j + 4 * i) * 4);
#pragma unroll
            for (int q = 0; q < 4; ++q) s += __expf(f[q] - m);
        }
        s += __shfl_xor(s, 1, 32);
        s += __shfl_xor(s, 2, 32);
        if (j == 0) rowInv[r] = 1.0f / s;
    }
    __syncthreads();

    // -------- Phase 2b: normalize -> scores (fp32) + P (bf16 in LDS) -------
    {
        bf16_t* ldsP = ldsQ;                     // Q region is dead now
#pragma unroll 4
        for (int chunk = tid; chunk < MROWS * (T_ENC / 4); chunk += 256) {
            const int r  = chunk >> 8;
            const int c4 = (chunk & 255) * 4;
            const float m   = funkey(rowMaxKey[r]);
            const float inv = rowInv[r];
            float* addr = scB + (size_t)r * T_ENC + c4;
            v4f f = *(const v4f*)addr;
            v4f p;
            v4bf pb;
#pragma unroll
            for (int q = 0; q < 4; ++q) {
                p[q]  = __expf(f[q] - m) * inv;
                pb[q] = (bf16_t)p[q];
            }
            __builtin_nontemporal_store(p, (v4f*)addr);   // final scores
            *(v4bf*)(ldsP + r * QP + c4) = pb;
        }
    }
    __syncthreads();

    // -------- Phase 3: O = P V (WMMA bf16) ---------------------------------
    {
        const bf16_t* arow = ldsQ + (mbase + ln16) * QP;  // P now
        float* Ob = summaries + ((size_t)batch * T_DEC + tb) * DIM;
        for (int nt = ngrp * 32; nt < ngrp * 32 + 32; ++nt) {
            const int nb = nt * 16;
            v8f acc = {};
            if constexpr (PRECONV) {
                const bf16_t* brow = VTb + (size_t)(nb + ln16) * T_ENC + 16 * hi;
#pragma unroll 2
                for (int k0 = 0; k0 < DIM; k0 += 32)
                    acc = __builtin_amdgcn_wmma_f32_16x16x32_bf16(
                            false, load_A_lds(arow + k0, hi),
                            false, load_B_bf16(brow + k0),
                            (short)0, acc, false, false);
            } else {
#pragma unroll 2
                for (int k0 = 0; k0 < DIM; k0 += 32)
                    acc = __builtin_amdgcn_wmma_f32_16x16x32_bf16(
                            false, load_A_lds(arow + k0, hi),
                            false, load_B_f32strided(
                                Vb + (size_t)(k0 + 16 * hi) * DIM + nb + ln16),
                            (short)0, acc, false, false);
            }
#pragma unroll
            for (int v = 0; v < 8; ++v)
                __builtin_nontemporal_store(acc[v],
                    Ob + (size_t)(mbase + v + 8 * hi) * DIM + nb + ln16);
        }
    }
}

extern "C" void kernel_launch(void* const* d_in, const int* in_sizes, int n_in,
                              void* d_out, int out_size, void* d_ws, size_t ws_size,
                              hipStream_t stream) {
    (void)in_sizes; (void)n_in; (void)out_size;
    const float* Q = (const float*)d_in[0];
    const float* K = (const float*)d_in[1];
    const float* V = (const float*)d_in[2];
    float* out       = (float*)d_out;
    float* summaries = out;                                   // [16,1024,1024]
    float* scores    = out + (size_t)NBATCH * T_DEC * DIM;    // [16,1024,1024]

    const size_t nElem = (size_t)NBATCH * T_ENC * DIM;        // 16M
    const size_t need  = 2 * nElem * sizeof(bf16_t);          // 64 MB
    const bool preconv = (ws_size >= need) && (d_ws != nullptr);

    dim3 grid(NBATCH * (T_DEC / MROWS));   // 256 workgroups
    dim3 block(256);                       // 8 wave32

    if (preconv) {
        bf16_t* Kbf  = (bf16_t*)d_ws;
        bf16_t* VTbf = Kbf + nElem;
        convert_k_bf16<<<dim3((unsigned)(nElem / (4 * 256))), dim3(256), 0, stream>>>(K, Kbf);
        transpose_v_bf16<<<dim3(DIM / 64, T_ENC / 64, NBATCH), dim3(256), 0, stream>>>(V, VTbf);
        DotAttention_75368086110276_kernel<true><<<grid, block, SMEM_TOTAL, stream>>>(
            Q, K, V, Kbf, VTbf, summaries, scores);
    } else {
        DotAttention_75368086110276_kernel<false><<<grid, block, SMEM_TOTAL, stream>>>(
            Q, K, V, nullptr, nullptr, summaries, scores);
    }
}
